// EfficientDeformableAttention_29970281792173
// MI455X (gfx1250) — matrix operand order repaired and checked by reading, not compiled, not run on hardware
//
#include <hip/hip_runtime.h>

#define DEV_INLINE __device__ __forceinline__

typedef __attribute__((ext_vector_type(16))) __bf16 v16bf;
typedef __attribute__((ext_vector_type(8)))  float  v8f;

union FragBits { v16bf v; uint4 q[2]; };

// Problem constants (fixed by the reference harness)
constexpr int Bn   = 4;
constexpr int NQ   = 16384;
constexpr int Cc   = 256;
constexpr int NHh  = 8;
constexpr int HD   = 32;      // Cc / NHh
constexpr int Hh   = 128;
constexpr int Ww   = 128;
constexpr int HWp  = Hh * Ww; // 16384
constexpr int Mtot = Bn * NQ; // 65536
constexpr int K    = 256;
constexpr int KTILES = 8;     // K / 32

DEV_INLINE unsigned short f2bf(float f) {
    unsigned u = __float_as_uint(f);
    u += 0x7fffu + ((u >> 16) & 1u);   // round-to-nearest-even
    return (unsigned short)(u >> 16);
}

// ---------------------------------------------------------------- converts
__global__ void k_f32_to_bf16(const float* __restrict__ src,
                              unsigned short* __restrict__ dst, int n) {
    int i = blockIdx.x * blockDim.x + threadIdx.x;
    if (i < n) dst[i] = f2bf(src[i]);
}

// ------------------------------------------------- pack W into B-fragment order
// Packed layout: flat index = (((nt*8 + kt)*32 + lane)*16 + j)
//   column n = nt*16 + (lane & 15)
//   k        = kt*32 + (lane >= 16 ? 16 : 0) + j
// Supports a two-source horizontal concat (W_off | W_attn).
__global__ void k_pack_B(const float* __restrict__ src0, int N0,
                         const float* __restrict__ src1, int Ntot,
                         unsigned short* __restrict__ dst) {
    int i = blockIdx.x * blockDim.x + threadIdx.x;
    int total = Ntot * K;
    if (i >= total) return;
    int j    = i & 15;
    int lane = (i >> 4) & 31;
    int kt   = (i >> 9) & 7;
    int nt   = i >> 12;
    int n = nt * 16 + (lane & 15);
    int k = kt * 32 + ((lane >> 4) << 4) + j;
    float w = (n < N0) ? src0[(size_t)k * N0 + n]
                       : src1[(size_t)k * (Ntot - N0) + (n - N0)];
    dst[i] = f2bf(w);
}

// ------------------------------------------------- core WMMA wave GEMM helper
// One wave computes 16 rows x (NT*16) cols of A[M,256] @ Bpacked, K=256.
// Software-pipelined: fragments for k-step kt+1 are fetched into a second
// register set while kt's WMMAs issue, so the scheduler can hoist the load
// clauses above the WMMA pack instead of serializing load->wait->wmma.
template <int NT>
DEV_INLINE void wave_gemm(const unsigned short* __restrict__ A,
                          const unsigned short* __restrict__ Bp,
                          int m0, int nt0, v8f (&acc)[NT]) {
    const int lane = threadIdx.x & 31;
    const int row  = m0 + (lane & 15);
    const int au   = (lane >> 4);                 // 0 or 1 -> +0 / +16 elements
    const uint4* Arow = (const uint4*)(A + (size_t)row * K);
    const uint4* Bb   = (const uint4*)Bp + ((size_t)nt0 * KTILES * 32 + lane) * 2;

    FragBits fa[2];
    FragBits fb[2][NT];

    auto loadA = [&](int kt, int s) {
        fa[s].q[0] = Arow[kt * 4 + au * 2];       // K = kt*32 + au*16 .. +7
        fa[s].q[1] = Arow[kt * 4 + au * 2 + 1];   // K = kt*32 + au*16 + 8 .. +15
    };
    auto loadB = [&](int kt, int s) {
#pragma unroll
        for (int t = 0; t < NT; ++t) {
            const uint4* p = Bb + ((size_t)(t * KTILES + kt)) * 64;
            fb[s][t].q[0] = p[0];
            fb[s][t].q[1] = p[1];
        }
    };

    loadA(0, 0);
    loadB(0, 0);
#pragma unroll
    for (int kt = 0; kt < KTILES; ++kt) {
        const int cur = kt & 1, nxt = cur ^ 1;
        if (kt + 1 < KTILES) {
            loadA(kt + 1, nxt);
            loadB(kt + 1, nxt);
        }
        __builtin_prefetch(Arow + (kt + 2) * 4, 0, 1);   // global_prefetch_b8
#pragma unroll
        for (int t = 0; t < NT; ++t) {
            acc[t] = __builtin_amdgcn_wmma_f32_16x16x32_bf16(
                false, fa[cur].v, false, fb[cur][t].v, (short)0, acc[t], false, false);
        }
    }
}

// ------------------------------------------------- GEMM: value -> v_proj (scattered)
// out layout: [b][h][hw][d]  (d contiguous)
__global__ void k_gemm_vproj(const unsigned short* __restrict__ A,
                             const unsigned short* __restrict__ Bp,
                             const float* __restrict__ bias,
                             float* __restrict__ vout) {
    const int wave = threadIdx.x >> 5, lane = threadIdx.x & 31;
    const int mi = wave & 3, ni = wave >> 2;
    const int m0  = blockIdx.x * 64 + mi * 16;
    const int nt0 = blockIdx.y * 8 + ni * 4;
    v8f acc[4] = {};
    wave_gemm<4>(A, Bp, m0, nt0, acc);
    const int madd = (lane >> 4) * 8;
#pragma unroll
    for (int t = 0; t < 4; ++t) {
        const int n = (nt0 + t) * 16 + (lane & 15);
        const float bv = bias[n];
        const int h = n >> 5, d = n & 31;
#pragma unroll
        for (int v = 0; v < 8; ++v) {
            const int m  = m0 + v + madd;
            const int b  = m >> 14;          // m / HWp
            const int hw = m & (HWp - 1);
            vout[((size_t)(b * NHh + h) * HWp + hw) * HD + d] = acc[t][v] + bv;
        }
    }
}

// ------------------------------------------------- GEMM: mid -> final output
__global__ void k_gemm_out(const unsigned short* __restrict__ A,
                           const unsigned short* __restrict__ Bp,
                           const float* __restrict__ bias,
                           float* __restrict__ out) {
    const int wave = threadIdx.x >> 5, lane = threadIdx.x & 31;
    const int mi = wave & 3, ni = wave >> 2;
    const int m0  = blockIdx.x * 64 + mi * 16;
    const int nt0 = blockIdx.y * 8 + ni * 4;
    v8f acc[4] = {};
    wave_gemm<4>(A, Bp, m0, nt0, acc);
    const int madd = (lane >> 4) * 8;
#pragma unroll
    for (int t = 0; t < 4; ++t) {
        const int n = (nt0 + t) * 16 + (lane & 15);
        const float bv = bias[n];
#pragma unroll
        for (int v = 0; v < 8; ++v) {
            const int m = m0 + v + madd;
            out[(size_t)m * Cc + n] = acc[t][v] + bv;
        }
    }
}

// ------------------------------------------------- GEMM: query -> [offsets|attn] raw (N=96)
__global__ void k_gemm96(const unsigned short* __restrict__ A,
                         const unsigned short* __restrict__ Bp,
                         const float* __restrict__ b_off,
                         const float* __restrict__ b_attn,
                         float* __restrict__ raw) {
    const int wave = threadIdx.x >> 5, lane = threadIdx.x & 31;
    const int m0 = blockIdx.x * 128 + wave * 16;
    v8f acc[6] = {};
    wave_gemm<6>(A, Bp, m0, 0, acc);
    const int madd = (lane >> 4) * 8;
#pragma unroll
    for (int t = 0; t < 6; ++t) {
        const int n = t * 16 + (lane & 15);
        const float bv = (n < 64) ? b_off[n] : b_attn[n - 64];
#pragma unroll
        for (int v = 0; v < 8; ++v) {
            const int m = m0 + v + madd;
            raw[(size_t)m * 96 + n] = acc[t][v] + bv;
        }
    }
}

// ------------------------------------------------- loc + softmax, in place
// row layout per (b,q): [64 floats: (h,p,2) offsets -> become (ix,iy)] [32: attn]
__global__ void k_loc_attn(const float* __restrict__ ref, float* __restrict__ la) {
    int i = blockIdx.x * blockDim.x + threadIdx.x;   // (b*NQ+q)*8 + h
    if (i >= Mtot * NHh) return;
    const int h  = i & 7;
    const int bq = i >> 3;
    const float rx = ref[(size_t)bq * 2 + 0];
    const float ry = ref[(size_t)bq * 2 + 1];
    float* row = la + (size_t)bq * 96;
    float ox[4], oy[4], aw[4];
#pragma unroll
    for (int p = 0; p < 4; ++p) {
        ox[p] = row[h * 8 + p * 2];
        oy[p] = row[h * 8 + p * 2 + 1];
        aw[p] = row[64 + h * 4 + p];
    }
    float mx = fmaxf(fmaxf(aw[0], aw[1]), fmaxf(aw[2], aw[3]));
    float s = 0.f;
#pragma unroll
    for (int p = 0; p < 4; ++p) { aw[p] = __expf(aw[p] - mx); s += aw[p]; }
    const float inv = 1.f / s;
#pragma unroll
    for (int p = 0; p < 4; ++p) {
        float lx = fminf(fmaxf(rx + ox[p] * (0.1f / (float)Ww), 0.f), 1.f);
        float ly = fminf(fmaxf(ry + oy[p] * (0.1f / (float)Hh), 0.f), 1.f);
        row[h * 8 + p * 2]     = lx * (float)Ww - 0.5f;   // ix
        row[h * 8 + p * 2 + 1] = ly * (float)Hh - 0.5f;   // iy
        row[64 + h * 4 + p]    = aw[p] * inv;
    }
}

// ------------------------------------------------- bilinear sample + attn-weighted sum
// One wave per (b,q,h); lane = head-dim channel d. Writes mid as bf16.
DEV_INLINE float corner(const float* __restrict__ vb, int x, int y, float w, int lane) {
    if ((unsigned)x < (unsigned)Ww && (unsigned)y < (unsigned)Hh)
        return w * vb[((size_t)(y * Ww + x)) * HD + lane];
    return 0.f;
}

__global__ void k_sample(const float* __restrict__ la,
                         const float* __restrict__ vproj,
                         unsigned short* __restrict__ mid) {
    const int gid  = blockIdx.x * blockDim.x + threadIdx.x;
    const int w    = gid >> 5;          // (b*NQ+q)*8 + h
    const int lane = gid & 31;
    const int h  = w & 7;
    const int bq = w >> 3;
    const int b  = bq >> 14;            // bq / NQ
    const float* row = la + (size_t)bq * 96;
    const float* vb  = vproj + (size_t)(b * NHh + h) * HWp * HD;
    float acc = 0.f;
#pragma unroll
    for (int p = 0; p < 4; ++p) {
        const float ix = row[h * 8 + p * 2];
        const float iy = row[h * 8 + p * 2 + 1];
        const float aw = row[64 + h * 4 + p];
        const float x0f = floorf(ix), y0f = floorf(iy);
        const float wx = ix - x0f, wy = iy - y0f;
        const int x0 = (int)x0f, y0 = (int)y0f;
        float s = 0.f;
        s += corner(vb, x0,     y0,     (1.f - wx) * (1.f - wy), lane);
        s += corner(vb, x0 + 1, y0,     wx * (1.f - wy),         lane);
        s += corner(vb, x0,     y0 + 1, (1.f - wx) * wy,         lane);
        s += corner(vb, x0 + 1, y0 + 1, wx * wy,                 lane);
        acc += aw * s;
    }
    mid[(size_t)bq * Cc + h * HD + lane] = f2bf(acc);
}

// ----------------------------------------------------------------- launcher
extern "C" void kernel_launch(void* const* d_in, const int* in_sizes, int n_in,
                              void* d_out, int out_size, void* d_ws, size_t ws_size,
                              hipStream_t stream) {
    (void)in_sizes; (void)n_in; (void)out_size; (void)ws_size;
    const float* query  = (const float*)d_in[0];
    const float* refp   = (const float*)d_in[1];
    const float* value  = (const float*)d_in[2];
    const float* W_off  = (const float*)d_in[3];
    const float* b_off  = (const float*)d_in[4];
    const float* W_attn = (const float*)d_in[5];
    const float* b_attn = (const float*)d_in[6];
    const float* W_v    = (const float*)d_in[7];
    const float* b_v    = (const float*)d_in[8];
    const float* W_out  = (const float*)d_in[9];
    const float* b_out  = (const float*)d_in[10];
    float* out = (float*)d_out;

    char* ws = (char*)d_ws;
    size_t cur = 0;
    auto alloc = [&](size_t bytes) {
        void* p = ws + cur;
        cur = (cur + bytes + 255) & ~(size_t)255;
        return p;
    };
    unsigned short* WvP   = (unsigned short*)alloc((size_t)256 * 256 * 2);
    unsigned short* WoutP = (unsigned short*)alloc((size_t)256 * 256 * 2);
    unsigned short* WoaP  = (unsigned short*)alloc((size_t)96 * 256 * 2);
    unsigned short* abuf  = (unsigned short*)alloc((size_t)Mtot * 256 * 2); // value_bf16 -> q_bf16 -> mid_bf16
    float* vproj = (float*)alloc((size_t)Mtot * 256 * 4);
    float* la    = (float*)alloc((size_t)Mtot * 96 * 4);

    // weight packing (tiny)
    k_pack_B<<<(256 * 256 + 255) / 256, 256, 0, stream>>>(W_v, 256, W_v, 256, WvP);
    k_pack_B<<<(256 * 256 + 255) / 256, 256, 0, stream>>>(W_out, 256, W_out, 256, WoutP);
    k_pack_B<<<(96 * 256 + 255) / 256, 256, 0, stream>>>(W_off, 64, W_attn, 96, WoaP);

    const int nElem = Mtot * 256;
    // value -> bf16 ; v_proj GEMM (WMMA)
    k_f32_to_bf16<<<nElem / 256, 256, 0, stream>>>(value, abuf, nElem);
    dim3 g256(Mtot / 64, 2);
    k_gemm_vproj<<<g256, 256, 0, stream>>>(abuf, WvP, b_v, vproj);

    // query -> bf16 ; offsets|attn GEMM (WMMA) ; softmax/loc
    k_f32_to_bf16<<<nElem / 256, 256, 0, stream>>>(query, abuf, nElem);
    k_gemm96<<<Mtot / 128, 256, 0, stream>>>(abuf, WoaP, b_off, b_attn, la);
    k_loc_attn<<<(Mtot * NHh) / 256, 256, 0, stream>>>(refp, la);

    // bilinear sampling -> mid (bf16, reuses abuf)
    k_sample<<<(Mtot * NHh * 32) / 256, 256, 0, stream>>>(la, vproj, abuf);

    // final projection GEMM (WMMA)
    k_gemm_out<<<g256, 256, 0, stream>>>(abuf, WoutP, b_out, out);
}
